// LSTM_Bahdanau_16990890622984
// MI455X (gfx1250) — matrix operand order repaired and checked by reading, not compiled, not run on hardware
//
#include <hip/hip_runtime.h>
#include <hip/hip_bf16.h>

// Problem constants (match reference)
#define B_  32
#define S_  64
#define T_  64
#define E_  512
#define H_  512
#define D_  1024
#define V_  32000

typedef __attribute__((ext_vector_type(16))) _Float16 v16h;
typedef __attribute__((ext_vector_type(8)))  _Float16 v8h;
typedef __attribute__((ext_vector_type(8)))  float    v8f;

// ---------------------------------------------------------------------------
// WMMA fragment loaders; A and B share one contiguous k order (not the ISA table order).
// A: 16(M)x32(K) f16. lanes 0-15 -> M=lane, lanes 16-31 -> M=lane-16.
//    elems 0..7  = K in [half*16, half*16+8)
//    elems 8..15 = K in [half*16+8, half*16+16)
// ---------------------------------------------------------------------------
__device__ __forceinline__ v16h wmma_load_a(const _Float16* __restrict__ p,
                                            int lda, int lane) {
  int m    = lane & 15;
  int half = lane >> 4;
  const _Float16* r = p + (long)m * lda + half * 16;
  v8h lo = *(const v8h*)(r);        // 16B
  v8h hi = *(const v8h*)(r + 8);    // 16B
  v16h out;
#pragma unroll
  for (int i = 0; i < 8; ++i) { out[i] = lo[i]; out[i + 8] = hi[i]; }
  return out;
}

// B fragment represents Bmat[K=32][N=16] with Bmat[k][n] = W[n][k] (W row-major
// [N,K]); lanes 0-15 -> n=lane, K=0..15; lanes 16-31 -> n=lane-16, K=16..31.
__device__ __forceinline__ v16h wmma_load_b(const _Float16* __restrict__ Wp,
                                            int ldw, int lane) {
  int n    = lane & 15;
  int half = lane >> 4;
  return *(const v16h*)(Wp + (long)n * ldw + half * 16);  // 32B contiguous
}

// C/D 16x16 f32 epilogue: lanes 0-15 n=lane rows M=r, lanes 16-31 rows M=r+8.
__device__ __forceinline__ void wmma_store_c(float* __restrict__ C, long ldc,
                                             const float* __restrict__ bias,
                                             const float* __restrict__ addend,
                                             long addStride, int mBase,
                                             int nBase, v8f c, int lane) {
  int n  = nBase + (lane & 15);
  int mb = mBase + ((lane >> 4) << 3);
#pragma unroll
  for (int r = 0; r < 8; ++r) {
    int   m = mb + r;
    float v = c[r];
    if (bias)   v += bias[n];
    if (addend) v += addend[(long)m * addStride + n];
    C[(long)m * ldc + n] = v;
  }
}

// ---------------------------------------------------------------------------
// Generic WMMA GEMM: C[M,N] = A[M,K] * W[N,K]^T (+ bias[N]) (+ addend[M,N]).
// A, W are f16 row-major; C is f32. Each wave computes one 32x32 output tile
// (2 A frags x 2 B frags -> 4 WMMAs per 32-wide K step), so for the M=32
// step-sequential decoder GEMMs each weight tile is fetched by exactly one
// wave. M must be a multiple of 32, N of 32, K of 32 (holds for all calls).
// ---------------------------------------------------------------------------
__global__ __launch_bounds__(256) void wmma_gemm_kernel(
    const _Float16* __restrict__ A, int lda,
    const _Float16* __restrict__ W, int ldw,
    float* __restrict__ C, long ldc,
    const float* __restrict__ bias,
    const float* __restrict__ addend, long addStride,
    int M, int N, int K) {
  int wid    = (int)((blockIdx.x * blockDim.x + threadIdx.x) >> 5);
  int tilesN = N >> 5;
  int total  = (M >> 5) * tilesN;
  if (wid >= total) return;                  // wave-uniform exit: EXEC stays all-1s
  int tm   = wid / tilesN;
  int tn   = wid - tm * tilesN;
  int lane = threadIdx.x & 31;

  const _Float16* A0 = A + (long)(tm * 32) * lda;
  const _Float16* A1 = A0 + (long)16 * lda;
  const _Float16* W0 = W + (long)(tn * 32) * ldw;
  const _Float16* W1 = W0 + (long)16 * ldw;

  v8f c00 = {}, c01 = {}, c10 = {}, c11 = {};
  for (int k = 0; k < K; k += 32) {
    v16h a0 = wmma_load_a(A0 + k, lda, lane);
    v16h a1 = wmma_load_a(A1 + k, lda, lane);
    v16h b0 = wmma_load_b(W0 + k, ldw, lane);
    v16h b1 = wmma_load_b(W1 + k, ldw, lane);
    c00 = __builtin_amdgcn_wmma_f32_16x16x32_f16(false, a0, false, b0, (short)0,
                                                 c00, false, false);
    c01 = __builtin_amdgcn_wmma_f32_16x16x32_f16(false, a0, false, b1, (short)0,
                                                 c01, false, false);
    c10 = __builtin_amdgcn_wmma_f32_16x16x32_f16(false, a1, false, b0, (short)0,
                                                 c10, false, false);
    c11 = __builtin_amdgcn_wmma_f32_16x16x32_f16(false, a1, false, b1, (short)0,
                                                 c11, false, false);
  }

  int m0 = tm * 32, n0 = tn * 32;
  wmma_store_c(C, ldc, bias, addend, addStride, m0,      n0,      c00, lane);
  wmma_store_c(C, ldc, bias, addend, addStride, m0,      n0 + 16, c01, lane);
  wmma_store_c(C, ldc, bias, addend, addStride, m0 + 16, n0,      c10, lane);
  wmma_store_c(C, ldc, bias, addend, addStride, m0 + 16, n0 + 16, c11, lane);
}

// ---------------------------------------------------------------------------
// LSTM pointwise: gates [Bn,4*Hn] f32 -> update c, h; emit f16 h for next GEMM
// and optionally scatter h into f16/f32 sequence buffers (pre-offset ptrs).
// ---------------------------------------------------------------------------
__global__ void lstm_cell_kernel(const float* __restrict__ gates,
                                 float* __restrict__ c,
                                 float* __restrict__ h32,
                                 _Float16* __restrict__ h16,
                                 _Float16* __restrict__ seq16,
                                 float* __restrict__ seq32,
                                 long seqStride, int Bn, int Hn) {
  int idx = blockIdx.x * blockDim.x + threadIdx.x;
  if (idx >= Bn * Hn) return;
  int b = idx / Hn, j = idx - b * Hn;
  const float* g = gates + (long)b * 4 * Hn;
  float ig = 1.f / (1.f + expf(-g[j]));
  float fg = 1.f / (1.f + expf(-g[Hn + j]));
  float gg = tanhf(g[2 * Hn + j]);
  float og = 1.f / (1.f + expf(-g[3 * Hn + j]));
  float cn = fg * c[idx] + ig * gg;
  float hn = og * tanhf(cn);
  c[idx]   = cn;
  h32[idx] = hn;
  h16[idx] = (_Float16)hn;
  if (seq16) seq16[(long)b * seqStride + j] = (_Float16)hn;
  if (seq32) seq32[(long)b * seqStride + j] = hn;
}

// f32 -> f16 weight conversion (grid-stride)
__global__ void f32_to_f16_kernel(const float* __restrict__ s,
                                  _Float16* __restrict__ d, long n) {
  long i      = (long)blockIdx.x * blockDim.x + threadIdx.x;
  long stride = (long)gridDim.x * blockDim.x;
  for (; i < n; i += stride) d[i] = (_Float16)s[i];
}

// emb_x[b,s,e] = emb_src[x[b, S-1-s], e]  (source sequence reversed)
__global__ void gather_src_kernel(const int* __restrict__ x,
                                  const float* __restrict__ emb,
                                  _Float16* __restrict__ out) {
  long idx = (long)blockIdx.x * blockDim.x + threadIdx.x;
  if (idx >= (long)B_ * S_ * E_) return;
  int  e   = (int)(idx & (E_ - 1));
  long bs  = idx >> 9;                 // /E_
  int  s   = (int)(bs & (S_ - 1));
  int  b   = (int)(bs >> 6);
  int  tok = x[b * S_ + (S_ - 1 - s)];
  out[idx] = (_Float16)emb[(long)tok * E_ + e];
}

// inp[b, 0:E] = emb_tgt[y[b,t], :]
__global__ void gather_tgt_kernel(const int* __restrict__ y,
                                  const float* __restrict__ emb,
                                  _Float16* __restrict__ inp, int t) {
  int idx = blockIdx.x * blockDim.x + threadIdx.x;
  if (idx >= B_ * E_) return;
  int b = idx / E_, e = idx - b * E_;
  int tok = y[b * T_ + t];
  inp[(long)b * (E_ + D_) + e] = (_Float16)emb[(long)tok * E_ + e];
}

// scores[b,s] = sum_e va[e]*tanh(q[b,e]+Uenc[b,s,e]);  w = softmax_s(scores)
__global__ __launch_bounds__(256) void attn_score_softmax_kernel(
    const float* __restrict__ q, const float* __restrict__ Uenc,
    const float* __restrict__ va, float* __restrict__ w) {
  int b    = blockIdx.x;
  int lane = threadIdx.x & 31;
  int wave = threadIdx.x >> 5;  // 8 waves
  __shared__ float sc[S_];
  const float* qb = q + (long)b * D_;
  for (int s = wave; s < S_; s += 8) {
    const float* u = Uenc + ((long)b * S_ + s) * D_;
    float acc = 0.f;
    for (int e = lane; e < D_; e += 32) acc += va[e] * tanhf(qb[e] + u[e]);
#pragma unroll
    for (int off = 16; off; off >>= 1) acc += __shfl_xor(acc, off, 32);
    if (lane == 0) sc[s] = acc;
  }
  __syncthreads();
  if (wave == 0) {
    float v0 = sc[lane], v1 = sc[lane + 32];
    float mx = fmaxf(v0, v1);
#pragma unroll
    for (int off = 16; off; off >>= 1) mx = fmaxf(mx, __shfl_xor(mx, off, 32));
    float e0 = expf(v0 - mx), e1 = expf(v1 - mx);
    float sum = e0 + e1;
#pragma unroll
    for (int off = 16; off; off >>= 1) sum += __shfl_xor(sum, off, 32);
    w[b * S_ + lane]      = e0 / sum;
    w[b * S_ + lane + 32] = e1 / sum;
  }
}

// inp[b, E + d] = sum_s w[b,s] * enc[b,s,d]
__global__ void attn_context_kernel(const float* __restrict__ w,
                                    const float* __restrict__ enc,
                                    _Float16* __restrict__ inp) {
  int idx = blockIdx.x * blockDim.x + threadIdx.x;
  if (idx >= B_ * D_) return;
  int b = idx >> 10, d = idx & (D_ - 1);
  float acc = 0.f;
  const float* wb = w + b * S_;
  const float* eb = enc + (long)b * S_ * D_ + d;
#pragma unroll 4
  for (int s = 0; s < S_; ++s) acc += wb[s] * eb[(long)s * D_];
  inp[(long)b * (E_ + D_) + E_ + d] = (_Float16)acc;
}

// decoder init: H0=cat(h0f,h0b), C0=cat(c0f,c0b), H1=cat(h1f,h1b), C1=...
__global__ void assemble_dec_state_kernel(
    const float* __restrict__ h0f, const float* __restrict__ c0f,
    const float* __restrict__ h0b, const float* __restrict__ c0b,
    const float* __restrict__ h1f, const float* __restrict__ c1f,
    const float* __restrict__ h1b, const float* __restrict__ c1b,
    float* __restrict__ H0, float* __restrict__ C0,
    float* __restrict__ H1, float* __restrict__ C1,
    _Float16* __restrict__ H0h, _Float16* __restrict__ H1h) {
  int idx = blockIdx.x * blockDim.x + threadIdx.x;
  if (idx >= B_ * H_) return;
  int b = idx / H_, j = idx - b * H_;
  long o0 = (long)b * D_ + j, o1 = o0 + H_;
  H0[o0] = h0f[idx];  H0[o1] = h0b[idx];
  C0[o0] = c0f[idx];  C0[o1] = c0b[idx];
  H1[o0] = h1f[idx];  H1[o1] = h1b[idx];
  C1[o0] = c1f[idx];  C1[o1] = c1b[idx];
  H0h[o0] = (_Float16)h0f[idx];  H0h[o1] = (_Float16)h0b[idx];
  H1h[o0] = (_Float16)h1f[idx];  H1h[o1] = (_Float16)h1b[idx];
}

// ------------------------------ host helpers -------------------------------
static inline void gemm(hipStream_t st, const _Float16* A, int lda,
                        const _Float16* W, int ldw, float* C, long ldc,
                        const float* bias, const float* addend, long addStride,
                        int M, int N, int K) {
  int tiles  = (M / 32) * (N / 32);
  int blocks = (tiles + 7) / 8;  // 8 waves / 256-thread block
  wmma_gemm_kernel<<<blocks, 256, 0, st>>>(A, lda, W, ldw, C, ldc, bias,
                                           addend, addStride, M, N, K);
}

static inline void lstm_cell(hipStream_t st, const float* gates, float* c,
                             float* h32, _Float16* h16, _Float16* seq16,
                             float* seq32, long seqStride, int Bn, int Hn) {
  int n = Bn * Hn;
  lstm_cell_kernel<<<(n + 255) / 256, 256, 0, st>>>(gates, c, h32, h16, seq16,
                                                    seq32, seqStride, Bn, Hn);
}

static inline void cvt16(hipStream_t st, const float* s, _Float16* d, long n) {
  int blocks = (int)((n + 1023) / 1024);
  if (blocks > 2048) blocks = 2048;
  f32_to_f16_kernel<<<blocks, 256, 0, st>>>(s, d, n);
}

extern "C" void kernel_launch(void* const* d_in, const int* in_sizes, int n_in,
                              void* d_out, int out_size, void* d_ws,
                              size_t ws_size, hipStream_t stream) {
  (void)in_sizes; (void)n_in; (void)out_size; (void)ws_size;
  const int*   x        = (const int*)  d_in[0];
  const int*   y        = (const int*)  d_in[1];
  const float* emb_src  = (const float*)d_in[2];
  const float* emb_tgt  = (const float*)d_in[3];
  const float* e0f_wih  = (const float*)d_in[4];
  const float* e0f_whh  = (const float*)d_in[5];
  const float* e0f_b    = (const float*)d_in[6];
  const float* e0b_wih  = (const float*)d_in[7];
  const float* e0b_whh  = (const float*)d_in[8];
  const float* e0b_b    = (const float*)d_in[9];
  const float* e1f_wih  = (const float*)d_in[10];
  const float* e1f_whh  = (const float*)d_in[11];
  const float* e1f_b    = (const float*)d_in[12];
  const float* e1b_wih  = (const float*)d_in[13];
  const float* e1b_whh  = (const float*)d_in[14];
  const float* e1b_b    = (const float*)d_in[15];
  const float* Wa_w     = (const float*)d_in[16];
  const float* Wa_b     = (const float*)d_in[17];
  const float* Ua_w     = (const float*)d_in[18];
  const float* Ua_b     = (const float*)d_in[19];
  const float* va_w     = (const float*)d_in[20];
  const float* d0_wih   = (const float*)d_in[21];
  const float* d0_whh   = (const float*)d_in[22];
  const float* d0_b     = (const float*)d_in[23];
  const float* d1_wih   = (const float*)d_in[24];
  const float* d1_whh   = (const float*)d_in[25];
  const float* d1_b     = (const float*)d_in[26];
  const float* fc_w     = (const float*)d_in[27];
  const float* fc_b     = (const float*)d_in[28];
  float*       out      = (float*)d_out;

  // -------- workspace bump allocator --------
  char*  ws  = (char*)d_ws;
  size_t off = 0;
  auto alloc = [&](size_t bytes) -> void* {
    void* p = ws + off;
    off += (bytes + 255) & ~(size_t)255;
    return p;
  };
  auto h16buf = [&](long n) { return (_Float16*)alloc((size_t)n * 2); };
  auto f32buf = [&](long n) { return (float*)alloc((size_t)n * 4); };

  // f16 weight copies (converted once per launch; deterministic)
  _Float16* w_e0f_ih = h16buf((long)4 * H_ * E_);
  _Float16* w_e0f_hh = h16buf((long)4 * H_ * H_);
  _Float16* w_e0b_ih = h16buf((long)4 * H_ * E_);
  _Float16* w_e0b_hh = h16buf((long)4 * H_ * H_);
  _Float16* w_e1f_ih = h16buf((long)4 * H_ * D_);
  _Float16* w_e1f_hh = h16buf((long)4 * H_ * H_);
  _Float16* w_e1b_ih = h16buf((long)4 * H_ * D_);
  _Float16* w_e1b_hh = h16buf((long)4 * H_ * H_);
  _Float16* w_Wa     = h16buf((long)D_ * D_);
  _Float16* w_Ua     = h16buf((long)D_ * D_);
  _Float16* w_d0_ih  = h16buf((long)4 * D_ * (E_ + D_));
  _Float16* w_d0_hh  = h16buf((long)4 * D_ * D_);
  _Float16* w_d1_ih  = h16buf((long)4 * D_ * D_);
  _Float16* w_d1_hh  = h16buf((long)4 * D_ * D_);
  _Float16* w_fc     = h16buf((long)V_ * D_);

  // activations
  _Float16* emb_x16 = h16buf((long)B_ * S_ * E_);
  float*    xg_f    = f32buf((long)B_ * S_ * 4 * H_);   // reused layer0 & layer1
  float*    xg_b    = f32buf((long)B_ * S_ * 4 * H_);
  _Float16* in1_16  = h16buf((long)B_ * S_ * D_);
  _Float16* enc16   = h16buf((long)B_ * S_ * D_);
  float*    enc32   = f32buf((long)B_ * S_ * D_);
  float*    Uenc    = f32buf((long)B_ * S_ * D_);

  // encoder per-direction state (h f32, h f16, c f32), 4 directions
  float*    st_h32[4]; _Float16* st_h16[4]; float* st_c[4];
  for (int d = 0; d < 4; ++d) {
    st_h32[d] = f32buf((long)B_ * H_);
    st_h16[d] = h16buf((long)B_ * H_);
    st_c[d]   = f32buf((long)B_ * H_);
  }
  float* gatesA = f32buf((long)B_ * 4 * H_);
  float* gatesB = f32buf((long)B_ * 4 * H_);

  // decoder state + temporaries
  float*    H0  = f32buf((long)B_ * D_);
  float*    C0  = f32buf((long)B_ * D_);
  float*    H1  = f32buf((long)B_ * D_);
  float*    C1  = f32buf((long)B_ * D_);
  _Float16* H0h = h16buf((long)B_ * D_);
  _Float16* H1h = h16buf((long)B_ * D_);
  float*    qb  = f32buf((long)B_ * D_);
  float*    attw   = f32buf((long)B_ * S_);
  _Float16* inp16  = h16buf((long)B_ * (E_ + D_));
  float*    dgate0 = f32buf((long)B_ * 4 * D_);
  float*    dgate1 = f32buf((long)B_ * 4 * D_);

  // -------- weight conversion --------
  cvt16(stream, e0f_wih, w_e0f_ih, (long)4 * H_ * E_);
  cvt16(stream, e0f_whh, w_e0f_hh, (long)4 * H_ * H_);
  cvt16(stream, e0b_wih, w_e0b_ih, (long)4 * H_ * E_);
  cvt16(stream, e0b_whh, w_e0b_hh, (long)4 * H_ * H_);
  cvt16(stream, e1f_wih, w_e1f_ih, (long)4 * H_ * D_);
  cvt16(stream, e1f_whh, w_e1f_hh, (long)4 * H_ * H_);
  cvt16(stream, e1b_wih, w_e1b_ih, (long)4 * H_ * D_);
  cvt16(stream, e1b_whh, w_e1b_hh, (long)4 * H_ * H_);
  cvt16(stream, Wa_w,    w_Wa,     (long)D_ * D_);
  cvt16(stream, Ua_w,    w_Ua,     (long)D_ * D_);
  cvt16(stream, d0_wih,  w_d0_ih,  (long)4 * D_ * (E_ + D_));
  cvt16(stream, d0_whh,  w_d0_hh,  (long)4 * D_ * D_);
  cvt16(stream, d1_wih,  w_d1_ih,  (long)4 * D_ * D_);
  cvt16(stream, d1_whh,  w_d1_hh,  (long)4 * D_ * D_);
  cvt16(stream, fc_w,    w_fc,     (long)V_ * D_);

  // -------- encoder --------
  {
    long n = (long)B_ * S_ * E_;
    gather_src_kernel<<<(int)((n + 255) / 256), 256, 0, stream>>>(x, emb_src,
                                                                  emb_x16);
  }
  // layer 0: xg = emb_x @ wih^T + b  (all timesteps at once)
  gemm(stream, emb_x16, E_, w_e0f_ih, E_, xg_f, 4 * H_, e0f_b, nullptr, 0,
       B_ * S_, 4 * H_, E_);
  gemm(stream, emb_x16, E_, w_e0b_ih, E_, xg_b, 4 * H_, e0b_b, nullptr, 0,
       B_ * S_, 4 * H_, E_);
  for (int d = 0; d < 2; ++d) {
    hipMemsetAsync(st_h32[d], 0, (size_t)B_ * H_ * 4, stream);
    hipMemsetAsync(st_h16[d], 0, (size_t)B_ * H_ * 2, stream);
    hipMemsetAsync(st_c[d],   0, (size_t)B_ * H_ * 4, stream);
  }
  for (int t = 0; t < S_; ++t) {
    int tb = S_ - 1 - t;
    gemm(stream, st_h16[0], H_, w_e0f_hh, H_, gatesA, 4 * H_, nullptr,
         xg_f + (long)t * 4 * H_, (long)S_ * 4 * H_, B_, 4 * H_, H_);
    gemm(stream, st_h16[1], H_, w_e0b_hh, H_, gatesB, 4 * H_, nullptr,
         xg_b + (long)tb * 4 * H_, (long)S_ * 4 * H_, B_, 4 * H_, H_);
    lstm_cell(stream, gatesA, st_c[0], st_h32[0], st_h16[0],
              in1_16 + (long)t * D_, nullptr, (long)S_ * D_, B_, H_);
    lstm_cell(stream, gatesB, st_c[1], st_h32[1], st_h16[1],
              in1_16 + (long)tb * D_ + H_, nullptr, (long)S_ * D_, B_, H_);
  }
  // layer 1 (input D): reuse xg buffers
  gemm(stream, in1_16, D_, w_e1f_ih, D_, xg_f, 4 * H_, e1f_b, nullptr, 0,
       B_ * S_, 4 * H_, D_);
  gemm(stream, in1_16, D_, w_e1b_ih, D_, xg_b, 4 * H_, e1b_b, nullptr, 0,
       B_ * S_, 4 * H_, D_);
  for (int d = 2; d < 4; ++d) {
    hipMemsetAsync(st_h32[d], 0, (size_t)B_ * H_ * 4, stream);
    hipMemsetAsync(st_h16[d], 0, (size_t)B_ * H_ * 2, stream);
    hipMemsetAsync(st_c[d],   0, (size_t)B_ * H_ * 4, stream);
  }
  for (int t = 0; t < S_; ++t) {
    int tb = S_ - 1 - t;
    gemm(stream, st_h16[2], H_, w_e1f_hh, H_, gatesA, 4 * H_, nullptr,
         xg_f + (long)t * 4 * H_, (long)S_ * 4 * H_, B_, 4 * H_, H_);
    gemm(stream, st_h16[3], H_, w_e1b_hh, H_, gatesB, 4 * H_, nullptr,
         xg_b + (long)tb * 4 * H_, (long)S_ * 4 * H_, B_, 4 * H_, H_);
    lstm_cell(stream, gatesA, st_c[2], st_h32[2], st_h16[2],
              enc16 + (long)t * D_, enc32 + (long)t * D_, (long)S_ * D_, B_, H_);
    lstm_cell(stream, gatesB, st_c[3], st_h32[3], st_h16[3],
              enc16 + (long)tb * D_ + H_, enc32 + (long)tb * D_ + H_,
              (long)S_ * D_, B_, H_);
  }
  // Uenc = enc @ Ua^T + Ua_b
  gemm(stream, enc16, D_, w_Ua, D_, Uenc, D_, Ua_b, nullptr, 0, B_ * S_, D_,
       D_);
  // decoder initial states
  assemble_dec_state_kernel<<<(B_ * H_ + 255) / 256, 256, 0, stream>>>(
      st_h32[0], st_c[0], st_h32[1], st_c[1], st_h32[2], st_c[2], st_h32[3],
      st_c[3], H0, C0, H1, C1, H0h, H1h);

  // -------- decoder --------
  for (int t = 0; t < T_ - 1; ++t) {
    // attention query: q = h1 @ Wa^T + Wa_b
    gemm(stream, H1h, D_, w_Wa, D_, qb, D_, Wa_b, nullptr, 0, B_, D_, D_);
    attn_score_softmax_kernel<<<B_, 256, 0, stream>>>(qb, Uenc, va_w, attw);
    attn_context_kernel<<<(B_ * D_ + 255) / 256, 256, 0, stream>>>(attw, enc32,
                                                                   inp16);
    gather_tgt_kernel<<<(B_ * E_ + 255) / 256, 256, 0, stream>>>(y, emb_tgt,
                                                                 inp16, t);
    // d0 cell: gates = inp @ wih^T + b + h0 @ whh^T
    gemm(stream, inp16, E_ + D_, w_d0_ih, E_ + D_, dgate0, 4 * D_, d0_b,
         nullptr, 0, B_, 4 * D_, E_ + D_);
    gemm(stream, H0h, D_, w_d0_hh, D_, dgate0, 4 * D_, nullptr, dgate0,
         4 * D_, B_, 4 * D_, D_);
    lstm_cell(stream, dgate0, C0, H0, H0h, nullptr, nullptr, 0, B_, D_);
    // d1 cell: gates = h0n @ wih^T + b + h1 @ whh^T   (H1h still previous h1)
    gemm(stream, H0h, D_, w_d1_ih, D_, dgate1, 4 * D_, d1_b, nullptr, 0, B_,
         4 * D_, D_);
    gemm(stream, H1h, D_, w_d1_hh, D_, dgate1, 4 * D_, nullptr, dgate1,
         4 * D_, B_, 4 * D_, D_);
    lstm_cell(stream, dgate1, C1, H1, H1h, nullptr, nullptr, 0, B_, D_);
    // logits[:, t, :] = h1n @ fc_w^T + fc_b
    gemm(stream, H1h, D_, w_fc, D_, out + (long)t * V_, (long)(T_ - 1) * V_,
         fc_b, nullptr, 0, B_, V_, D_);
  }
}
